// ScaleDotProductAttention_74388833567393
// MI455X (gfx1250) — compile-verified, matrix operand order checked by reading, not run on hardware
//
#include <hip/hip_runtime.h>
#include <hip/hip_bf16.h>

typedef __attribute__((ext_vector_type(16))) _Float16 v16h;
typedef __attribute__((ext_vector_type(8)))  _Float16 v8h;
typedef __attribute__((ext_vector_type(8)))  float    v8f;
typedef __attribute__((ext_vector_type(4)))  float    f32x4;
typedef __attribute__((ext_vector_type(4)))  int      v4i;

#if __has_builtin(__builtin_amdgcn_global_load_async_to_lds_b128) && \
    __has_builtin(__builtin_amdgcn_s_wait_asynccnt)
#define USE_ASYNC_LDS 1
#else
#define USE_ASYNC_LDS 0
#endif

namespace {
constexpr int   kB  = 2;
constexpr int   kH  = 16;
constexpr int   kT  = 2048;
constexpr int   kD  = 128;
constexpr int   kQB = 64;                       // query rows per block (4 waves x 16)
constexpr int   kWaves = kQB / 16;
constexpr float kScale = 0.08838834764831845f;  // 1/sqrt(128)

// element -> contraction index inside a 32-wide K slab (16-bit A/B fragment layout)
__device__ __forceinline__ int kmap(int e, int hi) {
  return (e & 7) + ((e >> 3) << 4) + (hi << 3);
}

#if USE_ASYNC_LDS
// async copy of one 32x128 f32 V slab (16KB): 128 threads x 8 x b128
__device__ __forceinline__ void stage_v_async(float* lds_dst, const float* gsrc,
                                              int tid) {
  #pragma unroll
  for (int u = 0; u < 8; ++u) {
    const int c = tid + 128 * u;  // 16B chunk index, consecutive tids coalesce
    __builtin_amdgcn_global_load_async_to_lds_b128(
        (__attribute__((address_space(1))) v4i*)(gsrc + c * 4),
        (__attribute__((address_space(3))) v4i*)(lds_dst + c * 4),
        0, 0);
  }
}
#else
__device__ __forceinline__ void stage_v_sync(float* lds_dst, const float* gsrc,
                                             int tid) {
  #pragma unroll
  for (int u = 0; u < 8; ++u) {
    const int c = tid + 128 * u;
    *(f32x4*)(lds_dst + c * 4) = *(const f32x4*)(gsrc + c * 4);
  }
}
#endif
}  // namespace

__global__ __launch_bounds__(kWaves * 32)
void sdpa_fused_wmma(const float* __restrict__ q, const float* __restrict__ k,
                     const float* __restrict__ v, float* __restrict__ out,
                     float* __restrict__ score) {
  __shared__ float    vraw[2][32 * 128];     // ping-pong raw f32 V slabs (async dst)
  __shared__ _Float16 vbufT[2][128][40];     // f16 V^T [d][s], padded rows
  __shared__ _Float16 pbuf[kWaves][16][40];  // per-wave P C-layout -> A-frag staging

  const int tid  = threadIdx.x;
  const int lane = tid & 31;
  const int wave = tid >> 5;
  const int hi   = lane >> 4;
  const int ln   = lane & 15;

  const int qtilesPerBH = kT / kQB;  // 32
  const int bh    = blockIdx.x / qtilesPerBH;
  const int qt    = blockIdx.x % qtilesPerBH;
  const int qrow0 = qt * kQB + wave * 16;

  const size_t bhBase = (size_t)bh * kT * kD;
  const float* qp = q + bhBase;
  const float* kp = k + bhBase;
  const float* vp = v + bhBase;
  float* op = out + bhBase + (size_t)qrow0 * kD;
  float* sp = score + (size_t)bh * kT * kT + (size_t)qrow0 * kT;

  // ---- load Q tile (16x128 f32) as 4 f16 A-fragments (16x32 each) ----
  v16h aq[4];
  {
    const float* qrow = qp + (size_t)(qrow0 + ln) * kD;
    #pragma unroll
    for (int c = 0; c < 4; ++c) {
      const int cb = c * 32 + hi * 8;
      f32x4 r0 = *(const f32x4*)(qrow + cb);
      f32x4 r1 = *(const f32x4*)(qrow + cb + 4);
      f32x4 r2 = *(const f32x4*)(qrow + cb + 16);
      f32x4 r3 = *(const f32x4*)(qrow + cb + 20);
      #pragma unroll
      for (int e = 0; e < 4; ++e) {
        aq[c][e]      = (_Float16)r0[e];
        aq[c][e + 4]  = (_Float16)r1[e];
        aq[c][e + 8]  = (_Float16)r2[e];
        aq[c][e + 12] = (_Float16)r3[e];
      }
    }
  }

  // ---- sweep 1: logits = (Q K^T)*scale -> score region; track row max ----
  float mx[8];
  #pragma unroll
  for (int r = 0; r < 8; ++r) mx[r] = -3.0e38f;

  for (int j = 0; j < kT / 16; ++j) {
    const float* krow = kp + (size_t)(j * 16 + ln) * kD;
    if (j + 1 < kT / 16)
      __builtin_prefetch(kp + (size_t)((j + 1) * 16 + ln) * kD, 0, 0);
    v8f acc = {0.f, 0.f, 0.f, 0.f, 0.f, 0.f, 0.f, 0.f};
    #pragma unroll
    for (int c = 0; c < 4; ++c) {
      const int cb = c * 32 + hi * 8;
      f32x4 r0 = *(const f32x4*)(krow + cb);
      f32x4 r1 = *(const f32x4*)(krow + cb + 4);
      f32x4 r2 = *(const f32x4*)(krow + cb + 16);
      f32x4 r3 = *(const f32x4*)(krow + cb + 20);
      v16h bk;
      #pragma unroll
      for (int e = 0; e < 4; ++e) {
        bk[e]      = (_Float16)r0[e];
        bk[e + 4]  = (_Float16)r1[e];
        bk[e + 8]  = (_Float16)r2[e];
        bk[e + 12] = (_Float16)r3[e];
      }
      acc = __builtin_amdgcn_wmma_f32_16x16x32_f16(false, aq[c], false, bk,
                                                   (short)0, acc, false, false);
    }
    #pragma unroll
    for (int r = 0; r < 8; ++r) {
      const float s = acc[r] * kScale;
      mx[r] = fmaxf(mx[r], s);
      sp[(size_t)(r + hi * 8) * kT + j * 16 + ln] = s;  // raw logits for now
    }
  }
  #pragma unroll
  for (int off = 1; off < 16; off <<= 1) {
    #pragma unroll
    for (int r = 0; r < 8; ++r)
      mx[r] = fmaxf(mx[r], __shfl_xor(mx[r], off, 32));
  }

  // ---- sweep 2: p = exp(s - m); row sums; O += P @ V (V staged via LDS) ----
  float ls[8];
  #pragma unroll
  for (int r = 0; r < 8; ++r) ls[r] = 0.f;
  v8f oacc[8];
  const v8f vzero = {0.f, 0.f, 0.f, 0.f, 0.f, 0.f, 0.f, 0.f};
  #pragma unroll
  for (int t = 0; t < 8; ++t) oacc[t] = vzero;

#if USE_ASYNC_LDS
  stage_v_async(vraw[0], vp, tid);  // prime the pipeline: slab 0 in flight
#endif

  for (int jp = 0; jp < kT / 32; ++jp) {
    const int buf = jp & 1;
#if USE_ASYNC_LDS
    if (jp + 1 < kT / 32)  // kick off next slab while we do softmax bookkeeping
      stage_v_async(vraw[(jp + 1) & 1], vp + (size_t)(jp + 1) * 32 * kD, tid);
#else
    stage_v_sync(vraw[buf], vp + (size_t)jp * 32 * kD, tid);
#endif

    // P for the two 16-col tiles of this slab (overlaps with async copy)
    #pragma unroll
    for (int jj = 0; jj < 2; ++jj) {
      const int j = jp * 2 + jj;
      #pragma unroll
      for (int r = 0; r < 8; ++r) {
        const size_t idx = (size_t)(r + hi * 8) * kT + j * 16 + ln;
        const float p = __expf(sp[idx] - mx[r]);
        ls[r] += p;
        sp[idx] = p;  // unnormalized; normalized in sweep 3 (L2-hot)
        pbuf[wave][r + hi * 8][jj * 16 + ln] = (_Float16)p;
      }
    }
    v16h pa;  // P as an A-fragment (per-wave LDS, in-wave ordering)
    #pragma unroll
    for (int e = 0; e < 16; ++e) pa[e] = pbuf[wave][ln][kmap(e, hi)];

#if USE_ASYNC_LDS
    if (jp + 1 < kT / 32)
      __builtin_amdgcn_s_wait_asynccnt(8);  // slab jp done; jp+1 may be in flight
    else
      __builtin_amdgcn_s_wait_asynccnt(0);
#endif
    __syncthreads();  // slab jp visible to whole block

    // transpose + f32->f16 convert: thread owns feature column d = tid
    {
      const float* col = &vraw[buf][tid];
      #pragma unroll
      for (int s0 = 0; s0 < 32; s0 += 8) {
        v8h pk;
        #pragma unroll
        for (int s = 0; s < 8; ++s) pk[s] = (_Float16)col[(s0 + s) * kD];
        *(v8h*)&vbufT[buf][tid][s0] = pk;
      }
    }
    __syncthreads();  // V^T f16 ready

    #pragma unroll
    for (int t = 0; t < 8; ++t) {
      const int d = t * 16 + ln;  // B-frag: lane holds column d, two b128 LDS reads
      v8h lo  = *(const v8h*)&vbufT[buf][d][hi * 8];
      v8h hi8 = *(const v8h*)&vbufT[buf][d][16 + hi * 8];
      v16h bv = __builtin_shufflevector(lo, hi8, 0, 1, 2, 3, 4, 5, 6, 7,
                                        8, 9, 10, 11, 12, 13, 14, 15);
      oacc[t] = __builtin_amdgcn_wmma_f32_16x16x32_f16(false, pa, false, bv,
                                                       (short)0, oacc[t], false, false);
    }
  }

  // finalize row sums and inverses
  #pragma unroll
  for (int off = 1; off < 16; off <<= 1) {
    #pragma unroll
    for (int r = 0; r < 8; ++r)
      ls[r] += __shfl_xor(ls[r], off, 32);
  }
  float linv[8];
  #pragma unroll
  for (int r = 0; r < 8; ++r) linv[r] = 1.0f / ls[r];

  // ---- write O = (P @ V) / l ----
  #pragma unroll
  for (int t = 0; t < 8; ++t) {
    #pragma unroll
    for (int r = 0; r < 8; ++r)
      op[(size_t)(r + hi * 8) * kD + t * 16 + ln] = oacc[t][r] * linv[r];
  }

  // ---- sweep 3: normalize score slice in place (resident in L2) ----
  for (int j = 0; j < kT / 16; ++j) {
    #pragma unroll
    for (int r = 0; r < 8; ++r) {
      const size_t idx = (size_t)(r + hi * 8) * kT + j * 16 + ln;
      sp[idx] *= linv[r];
    }
  }
}

extern "C" void kernel_launch(void* const* d_in, const int* in_sizes, int n_in,
                              void* d_out, int out_size, void* d_ws, size_t ws_size,
                              hipStream_t stream) {
  (void)in_sizes; (void)n_in; (void)out_size; (void)d_ws; (void)ws_size;
  const float* q = (const float*)d_in[0];
  const float* k = (const float*)d_in[1];
  const float* v = (const float*)d_in[2];
  float* out   = (float*)d_out;
  float* score = out + (size_t)kB * kH * kT * kD;  // tuple: (out, score) flat

  dim3 grid(kB * kH * (kT / kQB));   // 1024 blocks
  dim3 block(kWaves * 32);           // 128 threads = 4 wave32s
  sdpa_fused_wmma<<<grid, block, 0, stream>>>(q, k, v, out, score);
}
